// QuantizedLinear_43310450213551
// MI455X (gfx1250) — matrix operand order repaired and checked by reading, not compiled, not run on hardware
//
#include <hip/hip_runtime.h>
#include <hip/hip_bf16.h>

typedef __attribute__((ext_vector_type(16))) _Float16 v16h;
typedef __attribute__((ext_vector_type(8)))  float    v8f;
typedef int v4i_ __attribute__((vector_size(16)));   // matches builtin's int4 param

#define FP8_MAX 448.0f
#define QEPS 1e-8f

#if __has_builtin(__builtin_amdgcn_global_load_async_to_lds_b128)
#define HAVE_ASYNC_LDS 1
#endif

// async global -> LDS copy of 16 fp16 (32 bytes) as two b128 transfers
static __device__ __forceinline__ void copy16h_to_lds(const _Float16* gsrc, _Float16* ldst) {
#ifdef HAVE_ASYNC_LDS
    __builtin_amdgcn_global_load_async_to_lds_b128(
        (__attribute__((address_space(1))) v4i_*)gsrc,
        (__attribute__((address_space(3))) v4i_*)ldst, 0, 0);
    __builtin_amdgcn_global_load_async_to_lds_b128(
        (__attribute__((address_space(1))) v4i_*)(gsrc + 8),
        (__attribute__((address_space(3))) v4i_*)(ldst + 8), 0, 0);
#else
    *(uint4*)(ldst)     = *(const uint4*)(gsrc);
    *(uint4*)(ldst + 8) = *(const uint4*)(gsrc + 8);
#endif
}

static __device__ __forceinline__ void wait_async0() {
#ifdef HAVE_ASYNC_LDS
#if __has_builtin(__builtin_amdgcn_s_wait_asynccnt)
    __builtin_amdgcn_s_wait_asynccnt(0);
#else
    asm volatile("s_wait_asynccnt 0" ::: "memory");
#endif
#endif
}

// ---------------------------------------------------------------------------
// Kernel 1: weight quantization, one workgroup per 128x128 block of W (1024x1024)
// ---------------------------------------------------------------------------
__global__ __launch_bounds__(256) void wquant_kernel(const float* __restrict__ w,
                                                     _Float16* __restrict__ wq,
                                                     float* __restrict__ w_scale_sum) {
    __shared__ float red[256];
    const int bo = blockIdx.x >> 3;   // block row (O)
    const int bi = blockIdx.x & 7;    // block col (I)
    const int t  = threadIdx.x;

    float m = 0.0f;
#pragma unroll 4
    for (int j = 0; j < 64; ++j) {
        int idx = t + j * 256;            // 0..16383 within the 128x128 block
        int r = idx >> 7, c = idx & 127;
        float v = w[(size_t)(bo * 128 + r) * 1024 + bi * 128 + c];
        m = fmaxf(m, fabsf(v));
    }
    red[t] = m;
    __syncthreads();
    for (int s = 128; s > 0; s >>= 1) {
        if (t < s) red[t] = fmaxf(red[t], red[t + s]);
        __syncthreads();
    }
    const float bmax  = red[0];
    const float scale = (bmax > QEPS) ? (FP8_MAX / bmax) : 1.0f;

#pragma unroll 4
    for (int j = 0; j < 64; ++j) {
        int idx = t + j * 256;
        int r = idx >> 7, c = idx & 127;
        size_t g = (size_t)(bo * 128 + r) * 1024 + bi * 128 + c;
        float v = fminf(fmaxf(w[g] * scale, -FP8_MAX), FP8_MAX);
        wq[g] = (_Float16)v;
    }
    if (t == 0) atomicAdd(w_scale_sum, scale);
}

// ---------------------------------------------------------------------------
// Kernel 2: activation quantization. One workgroup per row (C=1024), one wave
// per 128-element tile, one lane handles 4 consecutive elements.
// ---------------------------------------------------------------------------
__global__ __launch_bounds__(256) void xquant_kernel(const float* __restrict__ x,
                                                     _Float16* __restrict__ xq,
                                                     float* __restrict__ x_scale_sum) {
    __shared__ float ssum[8];
    const int row  = blockIdx.x;
    const int wave = threadIdx.x >> 5;
    const int lane = threadIdx.x & 31;

    const size_t base = (size_t)row * 1024 + wave * 128 + lane * 4;
    float4 v = *(const float4*)(x + base);

    // match reference: values are bf16-rounded before max/quant
    float b0 = __bfloat162float(__float2bfloat16(v.x));
    float b1 = __bfloat162float(__float2bfloat16(v.y));
    float b2 = __bfloat162float(__float2bfloat16(v.z));
    float b3 = __bfloat162float(__float2bfloat16(v.w));

    float m = fmaxf(fmaxf(fabsf(b0), fabsf(b1)), fmaxf(fabsf(b2), fabsf(b3)));
#pragma unroll
    for (int off = 16; off > 0; off >>= 1)
        m = fmaxf(m, __shfl_xor(m, off, 32));

    const float scale = (m > QEPS) ? (FP8_MAX / m) : 1.0f;

    union { _Float16 h[4]; uint2 u; } pk;
    pk.h[0] = (_Float16)fminf(fmaxf(b0 * scale, -FP8_MAX), FP8_MAX);
    pk.h[1] = (_Float16)fminf(fmaxf(b1 * scale, -FP8_MAX), FP8_MAX);
    pk.h[2] = (_Float16)fminf(fmaxf(b2 * scale, -FP8_MAX), FP8_MAX);
    pk.h[3] = (_Float16)fminf(fmaxf(b3 * scale, -FP8_MAX), FP8_MAX);
    *(uint2*)(xq + base) = pk.u;

    if (lane == 0) ssum[wave] = scale;
    __syncthreads();
    if (threadIdx.x == 0) {
        float s = 0.0f;
#pragma unroll
        for (int i = 0; i < 8; ++i) s += ssum[i];
        atomicAdd(x_scale_sum, s);
    }
}

// ---------------------------------------------------------------------------
// Kernel 3: WMMA GEMM with double-buffered LDS fed by async global->LDS loads.
// out[M,N] = xq[M,K] * wq[N,K]^T, fp32 accumulation, then mean-scale dequant
// + bias, bf16 store.
// Block tile 128(M) x 128(N), BK = 32.  8 waves: 4(M) x 2(N), each wave
// computes 32x64 = 2x4 wmma tiles of 16x16.
// ---------------------------------------------------------------------------
__global__ __launch_bounds__(256) void gemm_kernel(const _Float16* __restrict__ A,  // [M,K] fp16
                                                   const _Float16* __restrict__ B,  // [N,K] fp16
                                                   const float* __restrict__ bias,
                                                   const float* __restrict__ sums,  // [x_sum, w_sum]
                                                   __hip_bfloat16* __restrict__ out,
                                                   int M) {
    const int K = 1024, N = 1024;
    __shared__ __align__(16) _Float16 lds_a[2][128 * 32];
    __shared__ __align__(16) _Float16 lds_b[2][128 * 32];

    const int bn = blockIdx.x;   // 0..7
    const int bm = blockIdx.y;   // 0..M/128-1
    const int t  = threadIdx.x;
    const int wave = t >> 5, lane = t & 31;
    const int wm = wave & 3;     // wave row in M (0..3)
    const int wn = wave >> 2;    // wave col in N (0..1)
    const int l16 = lane & 15, half = lane >> 4;

    v8f acc[2][4];
#pragma unroll
    for (int mi = 0; mi < 2; ++mi)
#pragma unroll
        for (int ni = 0; ni < 4; ++ni)
            acc[mi][ni] = (v8f){0.f, 0.f, 0.f, 0.f, 0.f, 0.f, 0.f, 0.f};

    // cooperative copy: thread t copies 16 halves: tile row = t>>1, chunk = (t&1)*16
    const int cr = t >> 1;
    const int cc = (t & 1) * 16;
    const _Float16* agp = A + (size_t)(bm * 128 + cr) * K + cc;
    const _Float16* bgp = B + (size_t)(bn * 128 + cr) * K + cc;
    const int loff = cr * 32 + cc;

    // stage K-step 0 into buffer 0
    copy16h_to_lds(agp, &lds_a[0][loff]);
    copy16h_to_lds(bgp, &lds_b[0][loff]);
    wait_async0();
    __syncthreads();

    int cur = 0;
    for (int kt = 32; kt <= K; kt += 32) {
        const bool more = (kt < K);
        if (more) {   // prefetch next K-step into the other buffer (overlaps WMMAs)
            copy16h_to_lds(agp + kt, &lds_a[cur ^ 1][loff]);
            copy16h_to_lds(bgp + kt, &lds_b[cur ^ 1][loff]);
        }

        // A fragments: lane<16 holds K {0..7,16..23}; lane>=16 holds K {8..15,24..31}
        v16h afr[2];
#pragma unroll
        for (int mi = 0; mi < 2; ++mi) {
            const int r  = wm * 32 + mi * 16 + l16;
            const int c0 = half ? 8 : 0;
            uint4* p = (uint4*)&afr[mi];
            p[0] = *(const uint4*)(&lds_a[cur][r * 32 + c0]);
            p[1] = *(const uint4*)(&lds_a[cur][r * 32 + c0 + 16]);
        }
        // B fragments: lane<16 holds K {0..7,16..23}; lane>=16 holds K {8..15,24..31}
        v16h bfr[4];
#pragma unroll
        for (int ni = 0; ni < 4; ++ni) {
            const int c  = wn * 64 + ni * 16 + l16;
            const int k0 = half ? 8 : 0;
            uint4* p = (uint4*)&bfr[ni];
            p[0] = *(const uint4*)(&lds_b[cur][c * 32 + k0]);
            p[1] = *(const uint4*)(&lds_b[cur][c * 32 + k0 + 16]);
        }

#pragma unroll
        for (int mi = 0; mi < 2; ++mi)
#pragma unroll
            for (int ni = 0; ni < 4; ++ni)
                acc[mi][ni] = __builtin_amdgcn_wmma_f32_16x16x32_f16(
                    false, afr[mi], false, bfr[ni],
                    (short)0, acc[mi][ni], false, false);

        if (more) {
            wait_async0();      // next buffer fully landed in LDS
            __syncthreads();    // and everyone finished reading the old one
            cur ^= 1;
        }
    }

    // global mean-scale dequant + bias
    const float x_mean = sums[0] / (float)(M * 8);
    const float w_mean = sums[1] * (1.0f / 64.0f);
    const float inv    = 1.0f / (x_mean * w_mean);

#pragma unroll
    for (int mi = 0; mi < 2; ++mi) {
#pragma unroll
        for (int ni = 0; ni < 4; ++ni) {
            const int colg = bn * 128 + wn * 64 + ni * 16 + l16;
            const float bv = bias[colg];
#pragma unroll
            for (int e = 0; e < 8; ++e) {
                const int rowg = bm * 128 + wm * 32 + mi * 16 + half * 8 + e;
                out[(size_t)rowg * N + colg] =
                    __float2bfloat16(acc[mi][ni][e] * inv + bv);
            }
        }
    }
}

// ---------------------------------------------------------------------------
// Launch
// ---------------------------------------------------------------------------
extern "C" void kernel_launch(void* const* d_in, const int* in_sizes, int n_in,
                              void* d_out, int out_size, void* d_ws, size_t ws_size,
                              hipStream_t stream) {
    const float* x    = (const float*)d_in[0];   // (8,4096,1024) fp32
    const float* w    = (const float*)d_in[1];   // (1024,1024) fp32
    const float* bias = (const float*)d_in[2];   // (1024,) fp32
    __hip_bfloat16* out = (__hip_bfloat16*)d_out;

    const int M = in_sizes[0] / 1024;            // 32768

    char*     ws   = (char*)d_ws;
    float*    sums = (float*)ws;                       // [0]=x_scale_sum, [1]=w_scale_sum
    _Float16* xq   = (_Float16*)(ws + 256);            // M*1024 fp16
    _Float16* wq   = (_Float16*)(ws + 256 + (size_t)M * 1024 * 2);  // 1024*1024 fp16

    (void)hipMemsetAsync(sums, 0, 2 * sizeof(float), stream);
    wquant_kernel<<<64, 256, 0, stream>>>(w, wq, sums + 1);
    xquant_kernel<<<M, 256, 0, stream>>>(x, xq, sums + 0);
    gemm_kernel<<<dim3(8, M / 128), 256, 0, stream>>>(xq, wq, bias, sums, out, M);
}
